// SelfAttention_15693810500376
// MI455X (gfx1250) — matrix operand order repaired and checked by reading, not compiled, not run on hardware
//
#include <hip/hip_runtime.h>
#include <hip/hip_bf16.h>

// ---------- CDNA5 WMMA / TDM types ----------
typedef __attribute__((ext_vector_type(16))) __bf16 v16bf;
typedef __attribute__((ext_vector_type(8)))  float  v8f;
typedef __attribute__((ext_vector_type(4)))  unsigned int u32x4;
typedef __attribute__((ext_vector_type(8)))  int          i32x8;
typedef __attribute__((ext_vector_type(4)))  int          i32x4;

union FragBF { v16bf v; uint4 u[2]; };

// fp32 -> bf16 round-to-nearest-even
__device__ __forceinline__ unsigned short f2bf(float f) {
    unsigned int u = __float_as_uint(f);
    unsigned int r = u + 0x7FFFu + ((u >> 16) & 1u);
    return (unsigned short)(r >> 16);
}
__device__ __forceinline__ float bf2f(unsigned short h) {
    return __uint_as_float(((unsigned int)h) << 16);
}
__device__ __forceinline__ void store_bf8(unsigned short* dst, const unsigned short h[8]) {
    uint4 u;
    u.x = (unsigned)h[0] | ((unsigned)h[1] << 16);
    u.y = (unsigned)h[2] | ((unsigned)h[3] << 16);
    u.z = (unsigned)h[4] | ((unsigned)h[5] << 16);
    u.w = (unsigned)h[6] | ((unsigned)h[7] << 16);
    *reinterpret_cast<uint4*>(dst) = u;
}

#define LDW 56   // LDS row stride in ushorts (112B: conflict-free, 16B aligned)

// ==========================================================================
// Kernel 1: QKV projection.  qkv[m,n] = sum_k x[b,k,s]*Win[k,n] + bin[n]
// M=32768 (m = b*4096+s), K=512, N=1536.  Block tile 64x128, K-step 32.
// Split-bf16 (hi+lo) WMMA, fp32 accumulate.
// ==========================================================================
__global__ __launch_bounds__(256) void qkv_gemm_kernel(
    const float* __restrict__ x, const float* __restrict__ Win,
    const float* __restrict__ bin, float* __restrict__ qkv)
{
    __shared__ __align__(16) unsigned short Ah[64 * LDW], Al[64 * LDW];
    __shared__ __align__(16) unsigned short Bh[128 * LDW], Bl[128 * LDW];

    const int t  = threadIdx.x;
    const int n0 = blockIdx.x * 128;
    const int m0 = blockIdx.y * 64;
    const int b  = m0 >> 12;           // 4096 tokens per batch image
    const int s0 = m0 & 4095;

    // A loader: thread -> (m in tile, k-group of 8). x is K-major: coalesced in s.
    const int a_m  = t & 63;
    const int a_kg = t >> 6;           // 0..3
    // B loader: thread -> (n in tile, k-group of 16). Coalesced in n.
    const int b_n  = t & 127;
    const int b_kg = t >> 7;           // 0..1

    const int w    = t >> 5;           // wave 0..7  (2 x 4 wave grid)
    const int l    = t & 31;
    const int wrow = (w & 1) * 32;
    const int wcol = (w >> 1) * 32;
    const int lr    = l & 15;
    const int koffA = (l & 16) ? 8 : 0;    // A frag: interleaved K halves
    const int koffB = (l & 16) ? 8 : 0;    // B frag: interleaved K halves

    v8f acc[2][2];
    #pragma unroll
    for (int i = 0; i < 2; ++i)
        #pragma unroll
        for (int j = 0; j < 2; ++j) acc[i][j] = (v8f)(0.0f);

    #pragma unroll 1
    for (int ks = 0; ks < 16; ++ks) {
        const int k0 = ks * 32;
        __syncthreads();
        // ---- stage A tile (64m x 32k) as bf16 hi/lo ----
        {
            const float* src = x + ((size_t)(b * 512 + k0 + a_kg * 8)) * 4096 + s0 + a_m;
            float f[8];
            #pragma unroll
            for (int j = 0; j < 8; ++j) f[j] = src[(size_t)j * 4096];
            unsigned short hi[8], lo[8];
            #pragma unroll
            for (int j = 0; j < 8; ++j) {
                hi[j] = f2bf(f[j]);
                lo[j] = f2bf(f[j] - bf2f(hi[j]));
            }
            store_bf8(&Ah[a_m * LDW + a_kg * 8], hi);
            store_bf8(&Al[a_m * LDW + a_kg * 8], lo);
        }
        // ---- stage B tile (32k x 128n) as bf16 hi/lo, LDS layout [n][k] ----
        {
            const float* src = Win + (size_t)(k0 + b_kg * 16) * 1536 + n0 + b_n;
            float f[16];
            #pragma unroll
            for (int j = 0; j < 16; ++j) f[j] = src[(size_t)j * 1536];
            unsigned short hi[16], lo[16];
            #pragma unroll
            for (int j = 0; j < 16; ++j) {
                hi[j] = f2bf(f[j]);
                lo[j] = f2bf(f[j] - bf2f(hi[j]));
            }
            store_bf8(&Bh[b_n * LDW + b_kg * 16],     hi);
            store_bf8(&Bh[b_n * LDW + b_kg * 16 + 8], hi + 8);
            store_bf8(&Bl[b_n * LDW + b_kg * 16],     lo);
            store_bf8(&Bl[b_n * LDW + b_kg * 16 + 8], lo + 8);
        }
        if (ks + 1 < 16) {  // prefetch next K tile (global_prefetch_b8)
            __builtin_prefetch(x + ((size_t)(b * 512 + k0 + 32 + a_kg * 8)) * 4096 + s0 + a_m, 0, 0);
            __builtin_prefetch(Win + (size_t)(k0 + 32 + b_kg * 16) * 1536 + n0 + b_n, 0, 0);
        }
        __syncthreads();

        // ---- load fragments ----
        FragBF ah[2], al[2], bh[2], bl[2];
        #pragma unroll
        for (int mt = 0; mt < 2; ++mt) {
            const int row = wrow + mt * 16 + lr;
            ah[mt].u[0] = *reinterpret_cast<const uint4*>(&Ah[row * LDW + koffA]);
            ah[mt].u[1] = *reinterpret_cast<const uint4*>(&Ah[row * LDW + 16 + koffA]);
            al[mt].u[0] = *reinterpret_cast<const uint4*>(&Al[row * LDW + koffA]);
            al[mt].u[1] = *reinterpret_cast<const uint4*>(&Al[row * LDW + 16 + koffA]);
        }
        #pragma unroll
        for (int nt = 0; nt < 2; ++nt) {
            const int col = wcol + nt * 16 + lr;
            bh[nt].u[0] = *reinterpret_cast<const uint4*>(&Bh[col * LDW + koffB]);
            bh[nt].u[1] = *reinterpret_cast<const uint4*>(&Bh[col * LDW + koffB + 16]);
            bl[nt].u[0] = *reinterpret_cast<const uint4*>(&Bl[col * LDW + koffB]);
            bl[nt].u[1] = *reinterpret_cast<const uint4*>(&Bl[col * LDW + koffB + 16]);
        }
        // ---- 12 WMMAs: hi*hi + hi*lo + lo*hi ----
        #pragma unroll
        for (int mt = 0; mt < 2; ++mt)
            #pragma unroll
            for (int nt = 0; nt < 2; ++nt) {
                acc[mt][nt] = __builtin_amdgcn_wmma_f32_16x16x32_bf16(
                    false, ah[mt].v, false, bh[nt].v, (short)0, acc[mt][nt], false, false);
                acc[mt][nt] = __builtin_amdgcn_wmma_f32_16x16x32_bf16(
                    false, ah[mt].v, false, bl[nt].v, (short)0, acc[mt][nt], false, false);
                acc[mt][nt] = __builtin_amdgcn_wmma_f32_16x16x32_bf16(
                    false, al[mt].v, false, bh[nt].v, (short)0, acc[mt][nt], false, false);
            }
    }

    // ---- epilogue: bias + store fp32 qkv[m, n] ----
    #pragma unroll
    for (int nt = 0; nt < 2; ++nt) {
        const int n  = n0 + wcol + nt * 16 + lr;
        const float bv = bin[n];
        #pragma unroll
        for (int mt = 0; mt < 2; ++mt) {
            #pragma unroll
            for (int r = 0; r < 8; ++r) {
                const int m = m0 + wrow + mt * 16 + r + ((l & 16) ? 8 : 0);
                qkv[(size_t)m * 1536 + n] = acc[mt][nt][r] + bv;
            }
        }
    }
}

// ==========================================================================
// Kernel 2: per-token attention over heads (8x8 softmax), one wave per token.
// Token staging done by the Tensor Data Mover: one 1D-tile descriptor copies
// 8 contiguous qkv rows (48 KB) global->LDS, tracked with TENSORcnt.
// ==========================================================================
__global__ __launch_bounds__(256) void head_attn_kernel(
    const float* __restrict__ qkv, float* __restrict__ ao)
{
    __shared__ __align__(16) float tok[8 * 1536];
    const int t = threadIdx.x;
    const size_t m0 = (size_t)blockIdx.x * 8;

    // ---- TDM stage: wave 0 issues tensor_load_to_lds (EXEC ignored by TDM) ----
    if ((t >> 5) == 0) {
        const unsigned long long ga =
            (unsigned long long)(uintptr_t)(const void*)(qkv + m0 * 1536);
        // low 32 bits of the generic shared pointer == hardware LDS byte offset
        const unsigned int lds_addr =
            (unsigned int)(uintptr_t)(void*)&tok[0];

        u32x4 g0;
        g0[0] = 1u;                                   // count=1 (valid user D#)
        g0[1] = lds_addr;                             // lds_addr [63:32]
        g0[2] = (unsigned int)(ga & 0xFFFFFFFFull);   // global_addr[31:0]
        g0[3] = (unsigned int)((ga >> 32) & 0x01FFFFFFull)
              | (2u << 30);                           // global_addr[56:32] | type=2

        i32x8 g1;
        g1[0] = (int)(2u << 16);             // workgroup_mask=0, data_size=4B
        g1[1] = (int)((12288u & 0xFFFFu) << 16);  // tensor_dim0[15:0] (bits 63:48)
        g1[2] = (int)(12288u >> 16);         // tensor_dim0[31:16]; tensor_dim1=0
        g1[3] = (int)(12288u << 16);         // tile_dim0=12288 (bits 127:112)
        g1[4] = 0;                           // tile_dim1=0, tile_dim2=0 (1D tile)
        g1[5] = 12288;                       // tensor_dim0_stride[31:0]
        g1[6] = 0;
        g1[7] = 0;

        const i32x4 gz4 = {0, 0, 0, 0};           // groups 2/3 unused (<=2D)
        const i32x8 gz8 = {0, 0, 0, 0, 0, 0, 0, 0};
        __builtin_amdgcn_tensor_load_to_lds(g0, g1, gz4, gz4, gz8, 0);
        __builtin_amdgcn_s_wait_tensorcnt(0);
    }
    __syncthreads();

    const int w = t >> 5;       // token within block
    const int l = t & 31;
    const float* tk = tok + w * 1536;

    const int i  = l >> 2;         // query head 0..7
    const int j0 = (l & 3) * 2;    // this lane owns key heads j0, j0+1

    const float* qp  = tk + i * 192;
    const float* kp0 = tk + j0 * 192 + 64;
    const float* kp1 = tk + (j0 + 1) * 192 + 64;
    float s0f = 0.f, s1f = 0.f;
    #pragma unroll
    for (int d4 = 0; d4 < 16; ++d4) {
        const float4 qv = *reinterpret_cast<const float4*>(qp  + d4 * 4);
        const float4 k0v = *reinterpret_cast<const float4*>(kp0 + d4 * 4);
        const float4 k1v = *reinterpret_cast<const float4*>(kp1 + d4 * 4);
        s0f += qv.x * k0v.x + qv.y * k0v.y + qv.z * k0v.z + qv.w * k0v.w;
        s1f += qv.x * k1v.x + qv.y * k1v.y + qv.z * k1v.z + qv.w * k1v.w;
    }
    s0f *= 0.125f;  // HEAD_DIM^-0.5
    s1f *= 0.125f;

    // softmax over 8 keys: row i lives in the 4-lane group (l>>2 == i)
    float mx = fmaxf(s0f, s1f);
    mx = fmaxf(mx, __shfl_xor(mx, 1));
    mx = fmaxf(mx, __shfl_xor(mx, 2));
    float e0 = __expf(s0f - mx), e1 = __expf(s1f - mx);
    float sm = e0 + e1;
    sm += __shfl_xor(sm, 1);
    sm += __shfl_xor(sm, 2);
    const float rinv = 1.0f / sm;
    const float a0 = e0 * rinv, a1 = e1 * rinv;

    // out[i, d] = sum_j attn[i,j] * v[j,d]; lane covers d = (l&3)*16 .. +15
    const int dseg = (l & 3) * 16;
    float ox[16];
    #pragma unroll
    for (int q = 0; q < 16; ++q) ox[q] = 0.f;
    #pragma unroll
    for (int jj = 0; jj < 4; ++jj) {
        const int srcl = i * 4 + jj;
        const float aj0 = __shfl(a0, srcl);
        const float aj1 = __shfl(a1, srcl);
        const float* vp0 = tk + (jj * 2) * 192 + 128 + dseg;
        const float* vp1 = tk + (jj * 2 + 1) * 192 + 128 + dseg;
        #pragma unroll
        for (int q = 0; q < 4; ++q) {
            const float4 v0 = *reinterpret_cast<const float4*>(vp0 + q * 4);
            const float4 v1 = *reinterpret_cast<const float4*>(vp1 + q * 4);
            ox[q * 4 + 0] += aj0 * v0.x + aj1 * v1.x;
            ox[q * 4 + 1] += aj0 * v0.y + aj1 * v1.y;
            ox[q * 4 + 2] += aj0 * v0.z + aj1 * v1.z;
            ox[q * 4 + 3] += aj0 * v0.w + aj1 * v1.w;
        }
    }
    float* op = ao + (m0 + w) * 512 + i * 64 + dseg;
    #pragma unroll
    for (int q = 0; q < 4; ++q) {
        float4 o4 = { ox[q * 4 + 0], ox[q * 4 + 1], ox[q * 4 + 2], ox[q * 4 + 3] };
        *reinterpret_cast<float4*>(op + q * 4) = o4;
    }
}

// ==========================================================================
// Kernel 3: output projection + transpose to (B, C, S).
// out[b,c,s] = sum_k ao[m,k]*Wout[k,c] + bout[c],  m = b*4096+s.
// Same WMMA scheme; output transposed through LDS for contiguous-S stores.
// ==========================================================================
__global__ __launch_bounds__(256) void out_gemm_kernel(
    const float* __restrict__ ao, const float* __restrict__ Wout,
    const float* __restrict__ bout, float* __restrict__ out)
{
    __shared__ __align__(16) unsigned short Ah[64 * LDW], Al[64 * LDW];
    __shared__ __align__(16) unsigned short Bh[128 * LDW], Bl[128 * LDW];
    __shared__ __align__(16) float OutT[128 * 68];   // [c][m], stride 272B

    const int t  = threadIdx.x;
    const int n0 = blockIdx.x * 128;   // c tile
    const int m0 = blockIdx.y * 64;
    const int b  = m0 >> 12;
    const int s0 = m0 & 4095;

    // A loader: ao is row-major [m][k] -> 4 threads cover one 32k row (coalesced)
    const int a_m  = t >> 2;
    const int a_kc = (t & 3) * 8;
    const int b_n  = t & 127;
    const int b_kg = t >> 7;

    const int w    = t >> 5;
    const int l    = t & 31;
    const int wrow = (w & 1) * 32;
    const int wcol = (w >> 1) * 32;
    const int lr    = l & 15;
    const int koffA = (l & 16) ? 8 : 0;
    const int koffB = (l & 16) ? 8 : 0;

    v8f acc[2][2];
    #pragma unroll
    for (int i = 0; i < 2; ++i)
        #pragma unroll
        for (int j = 0; j < 2; ++j) acc[i][j] = (v8f)(0.0f);

    #pragma unroll 1
    for (int ks = 0; ks < 16; ++ks) {
        const int k0 = ks * 32;
        __syncthreads();
        {   // A tile: 8 contiguous fp32 per thread
            const float* src = ao + (size_t)(m0 + a_m) * 512 + k0 + a_kc;
            float f[8];
            #pragma unroll
            for (int j = 0; j < 8; ++j) f[j] = src[j];
            unsigned short hi[8], lo[8];
            #pragma unroll
            for (int j = 0; j < 8; ++j) {
                hi[j] = f2bf(f[j]);
                lo[j] = f2bf(f[j] - bf2f(hi[j]));
            }
            store_bf8(&Ah[a_m * LDW + a_kc], hi);
            store_bf8(&Al[a_m * LDW + a_kc], lo);
        }
        {   // B tile: Wout row-major (K x 512), coalesced in n
            const float* src = Wout + (size_t)(k0 + b_kg * 16) * 512 + n0 + b_n;
            float f[16];
            #pragma unroll
            for (int j = 0; j < 16; ++j) f[j] = src[(size_t)j * 512];
            unsigned short hi[16], lo[16];
            #pragma unroll
            for (int j = 0; j < 16; ++j) {
                hi[j] = f2bf(f[j]);
                lo[j] = f2bf(f[j] - bf2f(hi[j]));
            }
            store_bf8(&Bh[b_n * LDW + b_kg * 16],     hi);
            store_bf8(&Bh[b_n * LDW + b_kg * 16 + 8], hi + 8);
            store_bf8(&Bl[b_n * LDW + b_kg * 16],     lo);
            store_bf8(&Bl[b_n * LDW + b_kg * 16 + 8], lo + 8);
        }
        if (ks + 1 < 16) {
            __builtin_prefetch(ao + (size_t)(m0 + a_m) * 512 + k0 + 32 + a_kc, 0, 0);
            __builtin_prefetch(Wout + (size_t)(k0 + 32 + b_kg * 16) * 512 + n0 + b_n, 0, 0);
        }
        __syncthreads();

        FragBF ah[2], al2[2], bh[2], bl2[2];
        #pragma unroll
        for (int mt = 0; mt < 2; ++mt) {
            const int row = wrow + mt * 16 + lr;
            ah[mt].u[0]  = *reinterpret_cast<const uint4*>(&Ah[row * LDW + koffA]);
            ah[mt].u[1]  = *reinterpret_cast<const uint4*>(&Ah[row * LDW + 16 + koffA]);
            al2[mt].u[0] = *reinterpret_cast<const uint4*>(&Al[row * LDW + koffA]);
            al2[mt].u[1] = *reinterpret_cast<const uint4*>(&Al[row * LDW + 16 + koffA]);
        }
        #pragma unroll
        for (int nt = 0; nt < 2; ++nt) {
            const int col = wcol + nt * 16 + lr;
            bh[nt].u[0]  = *reinterpret_cast<const uint4*>(&Bh[col * LDW + koffB]);
            bh[nt].u[1]  = *reinterpret_cast<const uint4*>(&Bh[col * LDW + koffB + 16]);
            bl2[nt].u[0] = *reinterpret_cast<const uint4*>(&Bl[col * LDW + koffB]);
            bl2[nt].u[1] = *reinterpret_cast<const uint4*>(&Bl[col * LDW + koffB + 16]);
        }
        #pragma unroll
        for (int mt = 0; mt < 2; ++mt)
            #pragma unroll
            for (int nt = 0; nt < 2; ++nt) {
                acc[mt][nt] = __builtin_amdgcn_wmma_f32_16x16x32_bf16(
                    false, ah[mt].v, false, bh[nt].v, (short)0, acc[mt][nt], false, false);
                acc[mt][nt] = __builtin_amdgcn_wmma_f32_16x16x32_bf16(
                    false, ah[mt].v, false, bl2[nt].v, (short)0, acc[mt][nt], false, false);
                acc[mt][nt] = __builtin_amdgcn_wmma_f32_16x16x32_bf16(
                    false, al2[mt].v, false, bh[nt].v, (short)0, acc[mt][nt], false, false);
            }
    }

    // ---- epilogue: bias + transpose via LDS ----
    __syncthreads();
    #pragma unroll
    for (int nt = 0; nt < 2; ++nt) {
        const int nloc = wcol + nt * 16 + lr;
        const float bv = bout[n0 + nloc];
        #pragma unroll
        for (int mt = 0; mt < 2; ++mt) {
            #pragma unroll
            for (int r = 0; r < 8; ++r) {
                const int mloc = wrow + mt * 16 + r + ((l & 16) ? 8 : 0);
                OutT[nloc * 68 + mloc] = acc[mt][nt][r] + bv;
            }
        }
    }
    __syncthreads();
    // writeback: each pair of threads streams one c-row (contiguous in s)
    {
        const int nloc = t >> 1;
        const int half = (t & 1) * 32;
        const int c = n0 + nloc;
        float* dptr = out + ((size_t)(b * 512 + c)) * 4096 + s0 + half;
        #pragma unroll
        for (int i = 0; i < 8; ++i) {
            const float4 vv = *reinterpret_cast<const float4*>(&OutT[nloc * 68 + half + i * 4]);
            *reinterpret_cast<float4*>(dptr + i * 4) = vv;
        }
    }
}

// ==========================================================================
extern "C" void kernel_launch(void* const* d_in, const int* in_sizes, int n_in,
                              void* d_out, int out_size, void* d_ws, size_t ws_size,
                              hipStream_t stream) {
    (void)in_sizes; (void)n_in; (void)out_size; (void)ws_size;
    const float* x    = (const float*)d_in[0];
    // d_in[1] is the unused _h scalar
    const float* Win  = (const float*)d_in[2];
    const float* bin  = (const float*)d_in[3];
    const float* Wout = (const float*)d_in[4];
    const float* bout = (const float*)d_in[5];
    float* out = (float*)d_out;

    float* qkv = (float*)d_ws;                         // 32768 x 1536 fp32 (192 MB)
    float* ao  = qkv + (size_t)32768 * 1536;           // 32768 x 512  fp32 (64 MB)

    qkv_gemm_kernel <<<dim3(12, 512), 256, 0, stream>>>(x, Win, bin, qkv);
    head_attn_kernel<<<4096,          256, 0, stream>>>(qkv, ao);
    out_gemm_kernel <<<dim3(4, 512),  256, 0, stream>>>(ao, Wout, bout, out);
}
